// MultiplicativeAttention_6640019440386
// MI455X (gfx1250) — matrix operand order repaired and checked by reading, not compiled, not run on hardware
//
#include <hip/hip_runtime.h>
#include <hip/hip_bf16.h>

typedef __attribute__((ext_vector_type(16))) _Float16 v16h;
typedef __attribute__((ext_vector_type(8)))  float    v8f;

#define D_MODEL 1024
#define D_K     128
#define SEQ     4096
#define BATCH   2
#define M_TOT   (BATCH * SEQ)     // 8192

static __device__ __forceinline__ v8f wmma_f16(const v16h& a, const v16h& b, const v8f& c) {
    return __builtin_amdgcn_wmma_f32_16x16x32_f16(false, a, false, b, (short)0, c, false, false);
}

static __device__ __forceinline__ v16h load_frag(const _Float16* p0, const _Float16* p1) {
    v16h f;
    uint4* fp = reinterpret_cast<uint4*>(&f);
    fp[0] = *reinterpret_cast<const uint4*>(p0);
    fp[1] = *reinterpret_cast<const uint4*>(p1);
    return f;
}

// Async 16B global -> LDS copy (ASYNCcnt-tracked). LDS offset = low 32 bits of
// the generic pointer.
static __device__ __forceinline__ void async_cp16(uint32_t lds_off, const void* gptr) {
    asm volatile("global_load_async_to_lds_b128 %0, %1, off"
                 :: "v"(lds_off), "v"((uint64_t)(uintptr_t)gptr) : "memory");
}
static __device__ __forceinline__ void wait_async0() {
    asm volatile("s_wait_asynccnt 0" ::: "memory");
}

// ---------------------------------------------------------------------------
// Kernel 0: f32 -> f16 convert of x (one pass, pure bandwidth).
// ---------------------------------------------------------------------------
__global__ __launch_bounds__(256) void cvt_kernel(const float* __restrict__ x,
                                                  _Float16* __restrict__ xh) {
    const size_t i = ((size_t)blockIdx.x * 256 + threadIdx.x) * 8;
    float4 a = *reinterpret_cast<const float4*>(x + i);
    float4 b = *reinterpret_cast<const float4*>(x + i + 4);
    _Float16 h[8] = {(_Float16)a.x, (_Float16)a.y, (_Float16)a.z, (_Float16)a.w,
                     (_Float16)b.x, (_Float16)b.y, (_Float16)b.z, (_Float16)b.w};
    *reinterpret_cast<uint4*>(xh + i) = *reinterpret_cast<const uint4*>(h);
}

// ---------------------------------------------------------------------------
// Kernel 1: projection. out = f16( xh[M,1024] @ w[1024,128] * scl ),
// optionally stored transposed (for V).  256 threads, 128x128 tile per block,
// double-buffered async x staging, K chunks of 32.
// ---------------------------------------------------------------------------
#define XROW 40   // padded row stride (halves) of x tile
#define WROW 40   // padded row stride (halves) of transposed w tile

__global__ __launch_bounds__(256) void proj_kernel(const _Float16* __restrict__ xh,
                                                   const float* __restrict__ w,
                                                   _Float16* __restrict__ outh,
                                                   float scl, int transpose_out) {
    __shared__ _Float16 Xt[2][128 * XROW];
    __shared__ _Float16 Wt[2][128 * WROW];

    const int t    = threadIdx.x;
    const int wave = t >> 5;
    const int lane = t & 31;
    const int half = lane >> 4;
    const int ln   = lane & 15;
    const int R0   = blockIdx.x * 128;

    const uint32_t xbase[2] = {(uint32_t)(uintptr_t)&Xt[0][0], (uint32_t)(uintptr_t)&Xt[1][0]};

    // ---- async stage of a 128x32 f16 x tile: 512 x 16B chunks ----
    auto issue_x = [&](int buf, int k0) {
#pragma unroll
        for (int i = 0; i < 2; ++i) {
            const int chunk = t + i * 256;
            const int row   = chunk >> 2;          // 4 chunks (64B) per row
            const int off   = (chunk & 3) * 16;    // bytes
            const char* g = (const char*)(xh + (size_t)(R0 + row) * D_MODEL + k0) + off;
            async_cp16(xbase[buf] + row * (XROW * 2) + off, g);
        }
    };
    // ---- synchronous stage of w chunk transposed (f32 -> f16) ----
    auto stage_w = [&](int buf, int k0) {
        const int k  = t >> 3;
        const int n0 = (t & 7) * 16;
        const float* gw = w + (size_t)(k0 + k) * D_K + n0;
#pragma unroll
        for (int i = 0; i < 16; ++i)
            Wt[buf][(n0 + i) * WROW + k] = (_Float16)gw[i];
    };

    v8f acc[8];
#pragma unroll
    for (int j = 0; j < 8; ++j) acc[j] = (v8f)0.0f;

    issue_x(0, 0);
    stage_w(0, 0);

    const int NCHUNK = D_MODEL / 32;               // 32
    for (int ki = 0; ki < NCHUNK; ++ki) {
        const int buf = ki & 1;
        wait_async0();
        __syncthreads();
        if (ki + 1 < NCHUNK) {
            issue_x(buf ^ 1, (ki + 1) * 32);
            stage_w(buf ^ 1, (ki + 1) * 32);
        }
        const int arow = wave * 16 + ln;
        v16h a = load_frag(&Xt[buf][arow * XROW + half * 8],
                           &Xt[buf][arow * XROW + half * 8 + 16]);
#pragma unroll
        for (int j = 0; j < 8; ++j) {
            const int n = j * 16 + ln;
            v16h bf = load_frag(&Wt[buf][n * WROW + half * 8],
                                &Wt[buf][n * WROW + half * 8 + 16]);
            acc[j] = wmma_f16(a, bf, acc[j]);
        }
    }

    // ---- epilogue (C layout: row = r + 8*half, col = j*16 + ln) ----
#pragma unroll
    for (int j = 0; j < 8; ++j) {
#pragma unroll
        for (int r = 0; r < 8; ++r) {
            const int row = R0 + wave * 16 + r + 8 * half;
            const int col = j * 16 + ln;
            const _Float16 hv = (_Float16)(acc[j][r] * scl);
            if (transpose_out)
                outh[(size_t)col * M_TOT + row] = hv;   // vT[n][b*4096+s]
            else
                outh[(size_t)row * D_K + col] = hv;
        }
    }
}

// ---------------------------------------------------------------------------
// Kernel 2: flash attention. 256 threads / 128 query rows per block.
// KV tiles of 32 rows, double-buffered async K / Vt staging.
// ---------------------------------------------------------------------------
#define KROW 136  // padded row stride (halves) of K tile  [32][128]
#define VROW 40   // padded row stride (halves) of Vt tile [128][32]
#define PROW 40   // padded row stride (halves) of P stage [16][32]

__global__ __launch_bounds__(256) void attn_kernel(const _Float16* __restrict__ qh,
                                                   const _Float16* __restrict__ kh,
                                                   const _Float16* __restrict__ vT,
                                                   float* __restrict__ out) {
    __shared__ _Float16 Kt[2][32 * KROW];
    __shared__ _Float16 Vt[2][128 * VROW];
    __shared__ _Float16 Ps[8 * 16 * PROW];

    const int t    = threadIdx.x;
    const int wave = t >> 5;
    const int lane = t & 31;
    const int half = lane >> 4;
    const int ln   = lane & 15;
    const int b    = blockIdx.y;
    const int m0   = blockIdx.x * 128 + wave * 16;

    const _Float16* qb = qh + (size_t)b * SEQ * D_K;
    const _Float16* kb = kh + (size_t)b * SEQ * D_K;

    const uint32_t kbase[2] = {(uint32_t)(uintptr_t)&Kt[0][0], (uint32_t)(uintptr_t)&Kt[1][0]};
    const uint32_t vbase[2] = {(uint32_t)(uintptr_t)&Vt[0][0], (uint32_t)(uintptr_t)&Vt[1][0]};

    auto issue_kv = [&](int buf, int t0) {
        // K tile: 32 rows x 256B = 512 x 16B chunks
#pragma unroll
        for (int i = 0; i < 2; ++i) {
            const int chunk = t + i * 256;
            const int row   = chunk >> 4;
            const int off   = (chunk & 15) * 16;
            const char* g = (const char*)(kb + (size_t)(t0 + row) * D_K) + off;
            async_cp16(kbase[buf] + row * (KROW * 2) + off, g);
        }
        // Vt tile: 128 rows x 64B = 512 x 16B chunks (vT rows are contiguous)
#pragma unroll
        for (int i = 0; i < 2; ++i) {
            const int chunk = t + i * 256;
            const int row   = chunk >> 2;
            const int off   = (chunk & 3) * 16;
            const char* g = (const char*)(vT + (size_t)row * M_TOT + (size_t)b * SEQ + t0) + off;
            async_cp16(vbase[buf] + row * (VROW * 2) + off, g);
        }
    };

    // Q fragments: 4 chunks of K=32 over d=128, A layout (scale pre-folded).
    v16h qf[4];
#pragma unroll
    for (int c = 0; c < 4; ++c) {
        const _Float16* qr = qb + (size_t)(m0 + ln) * D_K + c * 32 + half * 8;
        qf[c] = load_frag(qr, qr + 16);
    }

    v8f o[8];
#pragma unroll
    for (int j = 0; j < 8; ++j) o[j] = (v8f)0.0f;
    float mi[8], li[8];
#pragma unroll
    for (int r = 0; r < 8; ++r) { mi[r] = -__builtin_inff(); li[r] = 0.0f; }

    _Float16* Pw = &Ps[wave * 16 * PROW];

    issue_kv(0, 0);
    const int NT = SEQ / 32;                      // 128
    for (int ti = 0; ti < NT; ++ti) {
        const int buf = ti & 1;
        wait_async0();
        __syncthreads();
        if (ti + 1 < NT) issue_kv(buf ^ 1, (ti + 1) * 32);

        // ---- S = Q @ K^T (two 16x16 tiles) ----
        v8f s0 = (v8f)0.0f, s1 = (v8f)0.0f;
#pragma unroll
        for (int c = 0; c < 4; ++c) {
            const _Float16* b0 = &Kt[buf][(0 * 16 + ln) * KROW + c * 32 + half * 8];
            s0 = wmma_f16(qf[c], load_frag(b0, b0 + 16), s0);
        }
#pragma unroll
        for (int c = 0; c < 4; ++c) {
            const _Float16* b1 = &Kt[buf][(1 * 16 + ln) * KROW + c * 32 + half * 8];
            s1 = wmma_f16(qf[c], load_frag(b1, b1 + 16), s1);
        }

        // ---- online softmax ----
#pragma unroll
        for (int r = 0; r < 8; ++r) {
            float v0 = s0[r], v1 = s1[r];
            float mx = fmaxf(v0, v1);
            mx = fmaxf(mx, __shfl_xor(mx, 1, 32));
            mx = fmaxf(mx, __shfl_xor(mx, 2, 32));
            mx = fmaxf(mx, __shfl_xor(mx, 4, 32));
            mx = fmaxf(mx, __shfl_xor(mx, 8, 32));
            const float mnew  = fmaxf(mi[r], mx);
            const float alpha = __expf(mi[r] - mnew);
            const float p0 = __expf(v0 - mnew);
            const float p1 = __expf(v1 - mnew);
            float rs = p0 + p1;
            rs += __shfl_xor(rs, 1, 32);
            rs += __shfl_xor(rs, 2, 32);
            rs += __shfl_xor(rs, 4, 32);
            rs += __shfl_xor(rs, 8, 32);
            li[r] = li[r] * alpha + rs;
            mi[r] = mnew;
#pragma unroll
            for (int j = 0; j < 8; ++j) o[j][r] *= alpha;
            const int prow = r + 8 * half;
            Pw[prow * PROW + ln]      = (_Float16)p0;   // wave-private; LDS is
            Pw[prow * PROW + 16 + ln] = (_Float16)p1;   // in-order within a wave
        }

        // ---- O += P @ V ----
        v16h pf = load_frag(&Pw[ln * PROW + half * 8],
                            &Pw[ln * PROW + half * 8 + 16]);
#pragma unroll
        for (int j = 0; j < 8; ++j) {
            const _Float16* bv = &Vt[buf][(j * 16 + ln) * VROW + half * 8];
            o[j] = wmma_f16(pf, load_frag(bv, bv + 16), o[j]);
        }
    }

    // ---- finalize ----
#pragma unroll
    for (int r = 0; r < 8; ++r) {
        const float inv = 1.0f / li[r];
        const int row = m0 + r + 8 * half;
#pragma unroll
        for (int j = 0; j < 8; ++j)
            out[((size_t)b * SEQ + row) * D_K + j * 16 + ln] = o[j][r] * inv;
    }
}

// ---------------------------------------------------------------------------
extern "C" void kernel_launch(void* const* d_in, const int* in_sizes, int n_in,
                              void* d_out, int out_size, void* d_ws, size_t ws_size,
                              hipStream_t stream) {
    const float* x  = (const float*)d_in[0];
    const float* wq = (const float*)d_in[1];
    const float* wk = (const float*)d_in[2];
    const float* wv = (const float*)d_in[3];
    float* out = (float*)d_out;

    const size_t NX   = (size_t)M_TOT * D_MODEL;    // 8 Mi elements
    const size_t NQKV = (size_t)M_TOT * D_K;        // 1 Mi elements
    _Float16* xh = (_Float16*)d_ws;
    _Float16* qh = xh + NX;
    _Float16* kh = qh + NQKV;
    _Float16* vT = kh + NQKV;                       // [128][8192] transposed

    const float scale = 1.0f / 11.313708498984761f; // 1/sqrt(128)

    cvt_kernel<<<dim3(NX / (256 * 8)), 256, 0, stream>>>(x, xh);

    dim3 pgrid(M_TOT / 128);                        // 64 blocks
    proj_kernel<<<pgrid, 256, 0, stream>>>(xh, wq, qh, scale, 0);
    proj_kernel<<<pgrid, 256, 0, stream>>>(xh, wk, kh, 1.0f, 0);
    proj_kernel<<<pgrid, 256, 0, stream>>>(xh, wv, vT, 1.0f, 1);

    dim3 agrid(SEQ / 128, BATCH);                   // 32 x 2 blocks
    attn_kernel<<<agrid, 256, 0, stream>>>(qh, kh, vT, out);
}
